// MaskRCNNDeformableFeatureExtractor_10127532884573
// MI455X (gfx1250) — matrix operand order repaired and checked by reading, not compiled, not run on hardware
//
#include <hip/hip_runtime.h>
#include <hip/hip_bf16.h>

// ---- problem constants ----
#define PP      14
#define NBINS   (PP*PP)          // 196
#define SCALE_  0.0625f
#define TSTD    0.1f
#define CCH     256
#define HH      128
#define WW      128
#define NROIS   192
#define FC      1024
#define DFEAT   (CCH*NBINS)      // 50176

typedef __attribute__((ext_vector_type(16))) __bf16 v16bf;
typedef __attribute__((ext_vector_type(8)))  __bf16 v8bf;
typedef __attribute__((ext_vector_type(2)))  __bf16 v2bf;
typedef __attribute__((ext_vector_type(8)))  float  v8f;

// ------------------------------------------------------------------
// Transpose x [B,C,H,W] -> xt [B,H*W,C] so channel gathers coalesce.
// ------------------------------------------------------------------
__global__ void transpose_kernel(const float* __restrict__ x, float* __restrict__ xt) {
    __shared__ float tile[32][33];
    int b   = blockIdx.z;
    int hw0 = blockIdx.x * 32;
    int c0  = blockIdx.y * 32;
    int tx = threadIdx.x, ty = threadIdx.y;
    const float* xb  = x  + (size_t)b * CCH * (HH*WW);
    float*       xtb = xt + (size_t)b * (HH*WW) * CCH;
    #pragma unroll
    for (int i = 0; i < 32; i += 8)
        tile[ty + i][tx] = xb[(size_t)(c0 + ty + i) * (HH*WW) + hw0 + tx];
    __syncthreads();
    #pragma unroll
    for (int i = 0; i < 32; i += 8)
        xtb[(size_t)(hw0 + ty + i) * CCH + c0 + tx] = tile[tx][ty + i];
}

// ------------------------------------------------------------------
// Modulated deformable PS-RoI pooling. One block per (roi, bin),
// one thread per channel (taps are contiguous in C => coalesced).
// om == nullptr : zero offsets, write bf16 features to outB.
// om != nullptr : offsets/mask from om[192][588], write fp32 to outF.
// ------------------------------------------------------------------
__global__ void pool_kernel(const float* __restrict__ xt, const float* __restrict__ rois,
                            const float* __restrict__ om,
                            float* __restrict__ outF, __bf16* __restrict__ outB) {
    int blk = blockIdx.x;
    int r   = blk / NBINS;
    int bin = blk % NBINS;
    int ph  = bin / PP, pw = bin % PP;
    int c   = threadIdx.x;

    const float* R = rois + r * 5;
    int   b  = (int)R[0];
    float sw = rintf(R[1]) * SCALE_ - 0.5f;
    float sh = rintf(R[2]) * SCALE_ - 0.5f;
    float ew = (rintf(R[3]) + 1.f) * SCALE_ - 0.5f;
    float eh = (rintf(R[4]) + 1.f) * SCALE_ - 0.5f;
    float rw = fmaxf(ew - sw, 0.1f);
    float rh = fmaxf(eh - sh, 0.1f);
    float bw = rw * (1.f / PP), bh = rh * (1.f / PP);

    float tx = 0.f, ty = 0.f, mask = 1.f;
    if (om) {
        const float* o = om + (size_t)r * 588;
        tx   = o[bin] * TSTD;
        ty   = o[NBINS + bin] * TSTD;
        mask = 1.f / (1.f + __expf(-o[2 * NBINS + bin]));
    }
    float wstart = pw * bw + sw + tx * rw;
    float hstart = ph * bh + sh + ty * rh;

    const float* base = xt + (size_t)b * (HH*WW) * CCH;
    float sum = 0.f; int cnt = 0;
    #pragma unroll
    for (int ih = 0; ih < 2; ++ih) {
        #pragma unroll
        for (int iw = 0; iw < 2; ++iw) {
            float w = wstart + iw * (bw * 0.5f);
            float h = hstart + ih * (bh * 0.5f);
            bool valid = (w >= -0.5f) && (w <= WW - 0.5f) && (h >= -0.5f) && (h <= HH - 0.5f);
            if (!valid) continue;                       // uniform across block
            float wc = fminf(fmaxf(w, 0.f), (float)(WW - 1));
            float hc = fminf(fmaxf(h, 0.f), (float)(HH - 1));
            int x0 = (int)floorf(wc), y0 = (int)floorf(hc);
            int x1 = min(x0 + 1, WW - 1), y1 = min(y0 + 1, HH - 1);
            float lx = wc - x0, ly = hc - y0;
            float v00 = base[((size_t)(y0 * WW + x0)) * CCH + c];
            float v01 = base[((size_t)(y0 * WW + x1)) * CCH + c];
            float v10 = base[((size_t)(y1 * WW + x0)) * CCH + c];
            float v11 = base[((size_t)(y1 * WW + x1)) * CCH + c];
            sum += (1.f - ly) * ((1.f - lx) * v00 + lx * v01)
                 +        ly  * ((1.f - lx) * v10 + lx * v11);
            ++cnt;
        }
    }
    float val = sum / (float)max(cnt, 1);
    if (outB) outB[(size_t)r * DFEAT + (size_t)c * NBINS + bin] = (__bf16)val;
    if (outF) outF[((size_t)(r * CCH + c)) * NBINS + bin] = val * mask;
}

// ------------------------------------------------------------------
// Skinny GEMM: partial[split] = A[192,K(bf16)] x W[K,N(fp32->bf16)]
// Workgroup = 16-wide N slice x all 192 rows (w1 streamed exactly
// once across the device). 4 waves; wave w owns M-tiles {w,w+4,w+8}.
// K-step 64: B tile 64x16 staged fp32->bf16 in LDS ([n][k], 128B
// rows => 32B-aligned v16bf fragment reads). Each thread stages a
// k-adjacent row pair so conversions pack into v_cvt_pk_bf16_f32 +
// 4x ds_store_b32. 6 WMMAs per barrier pair. GUARD=true only for
// the N=588 tail GEMM.
// ------------------------------------------------------------------
template<bool GUARD>
__global__ void __launch_bounds__(128)
gemm_kernel(const __bf16* __restrict__ A, const float* __restrict__ W,
            float* __restrict__ partial,
            int K /*A row stride*/, int N, int Npad, int Kper) {
    __shared__ __attribute__((aligned(32))) __bf16 ldsB[16][64];   // [n][k]

    int ntile  = blockIdx.x;
    int split  = blockIdx.y;
    int k0base = split * Kper;
    int tid  = threadIdx.x;
    int lane = tid & 31;
    int wave = tid >> 5;

    int n_in = lane & 15;
    int kbA  = (lane >> 4) * 16;    // A frag: K = kbA..+15
    int kbB  = (lane >> 4) * 16;    // B frag: K = kbB..+15

    // cooperative B staging: thread owns rows k=2u,2u+1 x 4 n-columns
    int u  = tid >> 2;              // 0..31 -> k pair 2u (covers k 0..63)
    int n0 = (tid & 3) * 4;         // n in tile (0,4,8,12)
    int gn = ntile * 16 + n0;

    // strength-reduced streaming pointers
    const float*  wp  = W + (size_t)(k0base + 2 * u) * N + gn;
    const __bf16* pa0 = A + (size_t)((wave    ) * 16 + n_in) * K + k0base + kbA;
    const __bf16* pa1 = A + (size_t)((wave + 4) * 16 + n_in) * K + k0base + kbA;
    const __bf16* pa2 = A + (size_t)((wave + 8) * 16 + n_in) * K + k0base + kbA;

    v8f acc0 = {}, acc1 = {}, acc2 = {};

    for (int it = 0; it < Kper / 64; ++it) {
        // ---- stage 64x16 B tile (fp32 -> packed bf16, transposed) ----
        const float* wpH = wp + N;      // k = 2u+1
        float4 r0, r1;
        if (!GUARD || gn + 3 < N) {
            r0 = *(const float4*)wp;
            r1 = *(const float4*)wpH;
        } else {
            r0.x = (gn + 0 < N) ? wp[0]  : 0.f;
            r0.y = (gn + 1 < N) ? wp[1]  : 0.f;
            r0.z = (gn + 2 < N) ? wp[2]  : 0.f;
            r0.w = (gn + 3 < N) ? wp[3]  : 0.f;
            r1.x = (gn + 0 < N) ? wpH[0] : 0.f;
            r1.y = (gn + 1 < N) ? wpH[1] : 0.f;
            r1.z = (gn + 2 < N) ? wpH[2] : 0.f;
            r1.w = (gn + 3 < N) ? wpH[3] : 0.f;
        }
        v2bf p0; p0[0] = (__bf16)r0.x; p0[1] = (__bf16)r1.x;
        v2bf p1; p1[0] = (__bf16)r0.y; p1[1] = (__bf16)r1.y;
        v2bf p2; p2[0] = (__bf16)r0.z; p2[1] = (__bf16)r1.z;
        v2bf p3; p3[0] = (__bf16)r0.w; p3[1] = (__bf16)r1.w;
        __builtin_prefetch(wp + (size_t)64 * N, 0, 1);   // global_prefetch_b8
        __syncthreads();                 // prior-iteration fragment reads done
        *(v2bf*)&ldsB[n0 + 0][2 * u] = p0;   // packed ds_store_b32
        *(v2bf*)&ldsB[n0 + 1][2 * u] = p1;
        *(v2bf*)&ldsB[n0 + 2][2 * u] = p2;
        *(v2bf*)&ldsB[n0 + 3][2 * u] = p3;
        __syncthreads();                 // B tile visible

        // ---- B fragments: lane n_in, K=kbB..+15 / K=32+kbB..+15 ----
        v16bf bf0 = *(const v16bf*)&ldsB[n_in][kbB];
        v16bf bf1 = *(const v16bf*)&ldsB[n_in][kbB + 32];

        // ---- A fragments from global (L2-resident bf16) ----
        v8bf l00 = *(const v8bf*)(pa0);      v8bf h00 = *(const v8bf*)(pa0 + 8);
        v8bf l01 = *(const v8bf*)(pa0 + 32); v8bf h01 = *(const v8bf*)(pa0 + 40);
        v8bf l10 = *(const v8bf*)(pa1);      v8bf h10 = *(const v8bf*)(pa1 + 8);
        v8bf l11 = *(const v8bf*)(pa1 + 32); v8bf h11 = *(const v8bf*)(pa1 + 40);
        v8bf l20 = *(const v8bf*)(pa2);      v8bf h20 = *(const v8bf*)(pa2 + 8);
        v8bf l21 = *(const v8bf*)(pa2 + 32); v8bf h21 = *(const v8bf*)(pa2 + 40);
        v16bf a00, a01, a10, a11, a20, a21;
        #pragma unroll
        for (int i = 0; i < 8; ++i) {
            a00[i] = l00[i]; a00[i + 8] = h00[i];
            a01[i] = l01[i]; a01[i + 8] = h01[i];
            a10[i] = l10[i]; a10[i + 8] = h10[i];
            a11[i] = l11[i]; a11[i + 8] = h11[i];
            a20[i] = l20[i]; a20[i + 8] = h20[i];
            a21[i] = l21[i]; a21[i + 8] = h21[i];
        }
        acc0 = __builtin_amdgcn_wmma_f32_16x16x32_bf16(false, a00, false, bf0, (short)0, acc0, false, false);
        acc1 = __builtin_amdgcn_wmma_f32_16x16x32_bf16(false, a10, false, bf0, (short)0, acc1, false, false);
        acc2 = __builtin_amdgcn_wmma_f32_16x16x32_bf16(false, a20, false, bf0, (short)0, acc2, false, false);
        acc0 = __builtin_amdgcn_wmma_f32_16x16x32_bf16(false, a01, false, bf1, (short)0, acc0, false, false);
        acc1 = __builtin_amdgcn_wmma_f32_16x16x32_bf16(false, a11, false, bf1, (short)0, acc1, false, false);
        acc2 = __builtin_amdgcn_wmma_f32_16x16x32_bf16(false, a21, false, bf1, (short)0, acc2, false, false);

        wp  += (size_t)64 * N;
        pa0 += 64; pa1 += 64; pa2 += 64;
    }

    // ---- writeback: C/D layout -> partial[split][m][Npad] ----
    int rowoff = (lane >> 4) * 8;
    float* pbase = partial + (size_t)split * NROIS * Npad + (size_t)ntile * 16 + n_in;
    #pragma unroll
    for (int v = 0; v < 8; ++v) {
        pbase[(size_t)((wave    ) * 16 + rowoff + v) * Npad] = acc0[v];
        pbase[(size_t)((wave + 4) * 16 + rowoff + v) * Npad] = acc1[v];
        pbase[(size_t)((wave + 8) * 16 + rowoff + v) * Npad] = acc2[v];
    }
}

// ------------------------------------------------------------------
// Reduce split-K partials + bias (+ ReLU), emit fp32 and/or bf16.
// ------------------------------------------------------------------
__global__ void combine_kernel(const float* __restrict__ partial, const float* __restrict__ bias,
                               float* __restrict__ outF, __bf16* __restrict__ outB,
                               int N, int Npad, int KS, int relu) {
    int idx = blockIdx.x * blockDim.x + threadIdx.x;
    if (idx >= NROIS * N) return;
    int m = idx / N, n = idx % N;
    float s = bias[n];
    for (int k = 0; k < KS; ++k)
        s += partial[((size_t)k * NROIS + m) * Npad + n];
    if (relu) s = fmaxf(s, 0.f);
    if (outF) outF[idx] = s;
    if (outB) outB[idx] = (__bf16)s;
}

extern "C" void kernel_launch(void* const* d_in, const int* in_sizes, int n_in,
                              void* d_out, int out_size, void* d_ws, size_t ws_size,
                              hipStream_t stream) {
    const float* x    = (const float*)d_in[0];
    const float* rois = (const float*)d_in[1];
    const float* w1   = (const float*)d_in[2];
    const float* b1   = (const float*)d_in[3];
    const float* w2   = (const float*)d_in[4];
    const float* b2   = (const float*)d_in[5];
    const float* w3   = (const float*)d_in[6];
    const float* b3   = (const float*)d_in[7];
    float* out = (float*)d_out;

    char* ws = (char*)d_ws;
    size_t off = 0;
    auto alloc = [&](size_t bytes) -> void* {
        void* p = ws + off;
        off += (bytes + 255) & ~(size_t)255;
        return p;
    };
    float*  xt      = (float*) alloc((size_t)2 * HH * WW * CCH * 4);        // 32 MB
    __bf16* f0      = (__bf16*)alloc((size_t)NROIS * DFEAT * 2);            // 19 MB
    __bf16* h1      = (__bf16*)alloc((size_t)NROIS * FC * 2);
    __bf16* h2      = (__bf16*)alloc((size_t)NROIS * FC * 2);
    float*  om      = (float*) alloc((size_t)NROIS * 588 * 4);
    float*  partial = (float*) alloc((size_t)4 * NROIS * FC * 4);           // split-K max

    // 1. x -> xt  [B,H,W,C]
    transpose_kernel<<<dim3((HH*WW)/32, CCH/32, 2), dim3(32, 8), 0, stream>>>(x, xt);
    // 2. base pooling (zero offsets) -> bf16 features [192, 50176]
    pool_kernel<<<NROIS * NBINS, CCH, 0, stream>>>(xt, rois, nullptr, nullptr, f0);
    // 3. FC1: [192,50176]x[50176,1024], split-K=4 (w1 streamed exactly once)
    gemm_kernel<false><<<dim3(FC / 16, 4), 128, 0, stream>>>(f0, w1, partial, DFEAT, FC, FC, DFEAT / 4);
    combine_kernel<<<(NROIS * FC + 255) / 256, 256, 0, stream>>>(partial, b1, nullptr, h1, FC, FC, 4, 1);
    // 4. FC2: [192,1024]x[1024,1024]
    gemm_kernel<false><<<dim3(FC / 16, 1), 128, 0, stream>>>(h1, w2, partial, FC, FC, FC, FC);
    combine_kernel<<<(NROIS * FC + 255) / 256, 256, 0, stream>>>(partial, b2, nullptr, h2, FC, FC, 1, 1);
    // 5. FC3: [192,1024]x[1024,588] (N padded to 592 in partials)
    gemm_kernel<true><<<dim3(37, 1), 128, 0, stream>>>(h2, w3, partial, FC, 588, 592, FC);
    combine_kernel<<<(NROIS * 588 + 255) / 256, 256, 0, stream>>>(partial, b3, om, nullptr, 588, 592, 1, 0);
    // 6. deformable pooling with predicted offsets, x sigmoid(mask) -> out
    pool_kernel<<<NROIS * NBINS, CCH, 0, stream>>>(xt, rois, om, out, nullptr);
}
